// GroupAttention_two_dim_FC_68659347193907
// MI455X (gfx1250) — matrix operand order repaired and checked by reading, not compiled, not run on hardware
//
#include <hip/hip_runtime.h>
#include <math.h>

// CDNA5 / gfx1250: wave32, WMMA 16x16x32 f16->f32 (codegen-confirmed builtin).
typedef __attribute__((ext_vector_type(16))) _Float16 v16h;
typedef __attribute__((ext_vector_type(8)))  float    v8f;
typedef __attribute__((ext_vector_type(4)))  float    v4f;

// ---------------------------------------------------------------------------
// Kernel 1: LayerNorm row statistics (mu, 1/(sqrt(var)+eps)), unbiased var.
// One wave (32 lanes) per 256-wide row; 8 rows per 256-thread block.
// ---------------------------------------------------------------------------
__global__ __launch_bounds__(256) void ln_stats_kernel(
    const float* __restrict__ ctx, float2* __restrict__ stats, int rows) {
  int row  = blockIdx.x * 8 + (threadIdx.x >> 5);
  int lane = threadIdx.x & 31;
  if (row >= rows) return;
  const float* p = ctx + (size_t)row * 256 + lane * 8;
  float4 a = *(const float4*)p;
  float4 bq = *(const float4*)(p + 4);
  float vv[8] = {a.x, a.y, a.z, a.w, bq.x, bq.y, bq.z, bq.w};
  float s = 0.f;
#pragma unroll
  for (int i = 0; i < 8; ++i) s += vv[i];
  for (int o = 16; o; o >>= 1) s += __shfl_xor(s, o);
  float mu = s * (1.0f / 256.0f);
  float ss = 0.f;
#pragma unroll
  for (int i = 0; i < 8; ++i) { float d = vv[i] - mu; ss += d * d; }
  for (int o = 16; o; o >>= 1) ss += __shfl_xor(ss, o);
  float var = ss * (1.0f / 255.0f);          // mean(..)*(d/(d-1)) = ss/255
  float inv = 1.0f / (sqrtf(var) + 1e-6f);
  if (lane == 0) stats[row] = make_float2(mu, inv);
}

// ---------------------------------------------------------------------------
// Kernel 2: fused LayerNorm + flat-reinterpret 2x2 pooling -> ctx (B,1024,256).
// pooled flat index tid = b*262144 + a*1024 + bb*32 + e ; taps base,+1,+64,+65
// all lie in LN row b*4096 + a*16 + (bb>>1), channels c0,c0+1,c0+64,c0+65.
// ---------------------------------------------------------------------------
__global__ __launch_bounds__(256) void pool_kernel(
    const float* __restrict__ ctx, const float2* __restrict__ stats,
    const float* __restrict__ g, const float* __restrict__ bet,
    float* __restrict__ ctxp, int total) {
  int tid = blockIdx.x * blockDim.x + threadIdx.x;
  if (tid >= total) return;
  int b   = tid >> 18;
  int pp  = tid & 262143;
  int a   = pp >> 10;
  int rem = pp & 1023;
  int bb  = rem >> 5;
  int e   = rem & 31;
  int row = b * 4096 + a * 16 + (bb >> 1);
  int c0  = (bb & 1) * 128 + e * 2;
  size_t base = (size_t)row * 256 + c0;
  float2 lo = *(const float2*)(ctx + base);
  float2 hi = *(const float2*)(ctx + base + 64);
  float2 st = stats[row];
  float x0 = g[c0]      * (lo.x - st.x) * st.y + bet[c0];
  float x1 = g[c0 + 1]  * (lo.y - st.x) * st.y + bet[c0 + 1];
  float x2 = g[c0 + 64] * (hi.x - st.x) * st.y + bet[c0 + 64];
  float x3 = g[c0 + 65] * (hi.y - st.x) * st.y + bet[c0 + 65];
  ctxp[tid] = 0.25f * (x0 + x1 + x2 + x3);
}

// ---------------------------------------------------------------------------
// Kernel 3: gate computation. One wave32 block per instance
// (2 variants x B x 32). Row dots <C[k],w1>, <C[k],w2> via
// v_wmma_f32_16x16x32_f16 (A = 16 rows x K32 chunk of C, B = [w1|w2|0..]).
// Weights are staged to LDS once and all 8 B-fragments are built branchlessly
// before the tile loop, so the WMMA inner loop is pure ds_load/cvt/wmma.
// Prefix-sum of the w2-dots gives the causal-mean term in O(1) per (t,r).
// ---------------------------------------------------------------------------
__global__ __launch_bounds__(32) void gate_kernel(
    const float* __restrict__ ctxp, const float* __restrict__ fcw,
    const float* __restrict__ fcb, const float* __restrict__ prior_ij,
    const float* __restrict__ prior_xy, float* __restrict__ Mij,
    float* __restrict__ Mxy, float* __restrict__ gpre, int B) {
  __shared__ float ldsC[32 * 256];
  __shared__ float ldsW[512];    // fc_w staged (w1 | w2)
  __shared__ float ldsDot[64];   // [k][0]=<C[k],w1>, [k][1]=<C[k],w2>
  __shared__ float ldsPre[32];   // inclusive prefix of ldsDot[k][1]

  int lane = threadIdx.x;
  int inst = blockIdx.x;
  int var  = inst / (B * 32);    // 0 = ij (row blocks), 1 = xy (transposed)
  int m    = inst % (B * 32);
  int b    = m >> 5;
  int j    = m & 31;

  // Stage weights (512 floats, coalesced: 4 float4s per lane).
  {
    const float4* src = (const float4*)fcw;
    float4* dst = (float4*)ldsW;
    for (int i = lane; i < 128; i += 32) dst[i] = src[i];
  }
  // Stage the 32x256 row block into LDS (ij: one contiguous 32KB chunk).
  if (var == 0) {
    const float4* src = (const float4*)(ctxp + ((size_t)b * 1024 + (size_t)j * 32) * 256);
    float4* dst = (float4*)ldsC;
    for (int i = lane; i < 2048; i += 32) dst[i] = src[i];
  } else {
    for (int k = 0; k < 32; ++k) {
      const float4* src = (const float4*)(ctxp + ((size_t)b * 1024 + (size_t)k * 32 + j) * 256);
      float4* dst = (float4*)(ldsC + k * 256);
      for (int i = lane; i < 64; i += 32) dst[i] = src[i];
    }
  }
  __syncthreads();

  // Fragment layouts (16-bit, wave32):
  // A (16x32): lane<16 -> M=lane, K {0..15}; lane>=16 -> M=lane-16,
  //            K {16..31}.
  // B (32x16): VGPR v holds K {2v,2v+1} (lanes0-15, N=lane) and
  //            K {2v+16,2v+17} (lanes16-31, N=lane-16).
  int hf = (lane >> 4) & 1;
  int n  = lane & 15;

  // Precompute the 8 B fragments (one per K-chunk) — branchless selects.
  v16h Bfr[8];
#pragma unroll
  for (int c = 0; c < 8; ++c) {
    int kb = c * 32 + hf * 16;
#pragma unroll
    for (int v = 0; v < 8; ++v) {
      float wa0 = ldsW[kb + 2 * v];
      float wa1 = ldsW[kb + 2 * v + 1];
      float wb0 = ldsW[256 + kb + 2 * v];
      float wb1 = ldsW[256 + kb + 2 * v + 1];
      float b0 = (n == 0) ? wa0 : ((n == 1) ? wb0 : 0.f);
      float b1 = (n == 0) ? wa1 : ((n == 1) ? wb1 : 0.f);
      Bfr[c][2 * v]     = (_Float16)b0;
      Bfr[c][2 * v + 1] = (_Float16)b1;
    }
  }

  // WMMA row-dot GEMM: two 16-row tiles, K=256 in 8 chunks of 32.
  for (int t2 = 0; t2 < 2; ++t2) {
    v8f acc = {};
    int row = n + t2 * 16;
    int ka0 = hf * 16;
#pragma unroll
    for (int c = 0; c < 8; ++c) {
      v16h Af;
      int ka = c * 32 + ka0;
#pragma unroll
      for (int e = 0; e < 8; ++e) {
        Af[e]     = (_Float16)ldsC[row * 256 + ka + e];
        Af[e + 8] = (_Float16)ldsC[row * 256 + ka + 8 + e];
      }
      acc = __builtin_amdgcn_wmma_f32_16x16x32_f16(
          false, Af, false, Bfr[c], (short)0, acc, false, false);
    }
    // D layout: VGPR q -> (M = q + (lane>=16 ? 8 : 0) + 16*t2, N = lane&15).
    if (n < 2) {
      int baseM = t2 * 16 + hf * 8;
#pragma unroll
      for (int q = 0; q < 8; ++q) ldsDot[(baseM + q) * 2 + n] = acc[q];
    }
  }
  __syncthreads();

  if (lane == 0) {
    float run = 0.f;
    for (int k = 0; k < 32; ++k) { run += ldsDot[k * 2 + 1]; ldsPre[k] = run; }
  }
  __syncthreads();

  float fb = fcb[0];
  float pr = (var == 0) ? prior_ij[0] : prior_xy[0];
  for (int idx = lane; idx < 1024; idx += 32) {
    int t = idx >> 5, r = idx & 31;
    float logit;
    if (r < t) {
      logit = fb;  // mean mask empty: mean==0 everywhere -> c1 zeroed too
    } else {
      float sp = (t > 0) ? ldsPre[t - 1] : 0.f;
      float mterm = (ldsPre[r] - sp) / (float)(r - t + 1);
      logit = ldsDot[t * 2 + 0] + mterm + fb;
    }
    float gv = 1.f / (1.f + expf(-logit));
    size_t mo = (size_t)b * 32768 + (size_t)j * 1024 + idx;
    size_t go = (size_t)(var * B + b) * 32768 + (size_t)j * 1024 + idx;
    if (var == 0) {
      float Mv = pr + (1.f - pr) * gv;
      Mij[mo] = Mv;
      gpre[go] = (t > r) ? 1.f : ((Mv == 0.f) ? 1.f : Mv);
    } else {
      float mt = (t <= r) ? gv : 0.f;
      float Mv = pr + (1.f - pr) * mt;
      Mxy[mo] = Mv;
      gpre[go] = (Mv == 0.f) ? 1.f : Mv;
    }
  }
}

// ---------------------------------------------------------------------------
// Kernel 4: cumprod along t. One thread per (variant*b, i, r), serial in t.
// ---------------------------------------------------------------------------
__global__ void cumprod_kernel(const float* __restrict__ gpre,
                               float* __restrict__ pbuf, int total) {
  int tid = blockIdx.x * blockDim.x + threadIdx.x;
  if (tid >= total) return;
  int vb = tid >> 10;
  int i  = (tid >> 5) & 31;
  int r  = tid & 31;
  size_t base = (size_t)vb * 32768 + (size_t)i * 1024 + r;
  float run = 1.f;
  for (int t = 0; t < 32; ++t) {
    run *= gpre[base + t * 32];
    pbuf[base + t * 32] = run;
  }
}

// ---------------------------------------------------------------------------
// Kernel 5: finalize: G[i,t,r] = (t==r) ? 1 : T(i,t,r)+T(t,i,r),
//           T(x,y,r) = (y<=r) ? P[x,y,r] : 0   (the swapaxes(1,2) mix).
// ---------------------------------------------------------------------------
__global__ void combine_kernel(const float* __restrict__ pbuf,
                               float* __restrict__ gfin, int total) {
  int tid = blockIdx.x * blockDim.x + threadIdx.x;
  if (tid >= total) return;
  int vb  = tid >> 15;
  int rem = tid & 32767;
  int i = rem >> 10;
  int t = (rem >> 5) & 31;
  int r = rem & 31;
  float val;
  if (t == r) {
    val = 1.f;
  } else {
    size_t vbb = (size_t)vb * 32768;
    float t1 = (t <= r) ? pbuf[vbb + (size_t)i * 1024 + t * 32 + r] : 0.f;
    float t2 = (i <= r) ? pbuf[vbb + (size_t)t * 1024 + i * 32 + r] : 0.f;
    val = t1 + t2;
  }
  gfin[tid] = val;
}

// ---------------------------------------------------------------------------
// Kernel 6: expansion (the bandwidth-dominant 134MB store).
// out[b, a2*64+r2, c2*64+e2] = Gij[b, c, a, r] * Gxy[b, a, c, e],
//   a=a2&31, r=r2>>1, c=c2>>1, e=e2>>1. One block per (b,a2): stages the
// two 32x32 tables in LDS, then streams 64x4096 floats as coalesced,
// non-temporal float4 stores (write-once data; keep it out of caches).
// ---------------------------------------------------------------------------
__global__ __launch_bounds__(256) void expand_kernel(
    const float* __restrict__ gfin, float* __restrict__ up, int B) {
  __shared__ float gij[1024];  // [c][r] = Gij_final[b, c, a, r]
  __shared__ float gxy[1024];  // [c][e] = Gxy_final[b, a, c, e]
  int tid = threadIdx.x;
  int blk = blockIdx.x;
  int b  = blk >> 6;
  int a2 = blk & 63;
  int a  = a2 & 31;
  const float* gijSrc = gfin + (size_t)b * 32768;                       // var 0
  const float* gxySrc = gfin + (size_t)(B + b) * 32768 + (size_t)a * 1024;
  for (int i = tid; i < 1024; i += 256) {
    int c = i >> 5, r = i & 31;
    gij[i] = gijSrc[(size_t)c * 1024 + a * 32 + r];
    gxy[i] = gxySrc[i];
  }
  __syncthreads();
  v4f* outBase = (v4f*)(up + (size_t)b * 16777216 + (size_t)a2 * 64 * 4096);
  for (int i = tid; i < 65536; i += 256) {   // 64 rows x 1024 float4/row
    int r2 = i >> 10;
    int p4 = i & 1023;
    int p  = p4 << 2;
    int c2 = p >> 6;
    int e2 = p & 63;
    int c = c2 >> 1;
    int e = e2 >> 1;
    int r = r2 >> 1;
    float gi = gij[c * 32 + r];
    float g0 = gi * gxy[c * 32 + e];
    float g1 = gi * gxy[c * 32 + e + 1];
    v4f outv = {g0, g0, g1, g1};
    __builtin_nontemporal_store(outv, &outBase[(size_t)r2 * 1024 + p4]);
  }
}

// ---------------------------------------------------------------------------
// Host launcher.
// Inputs: context, mask(unused), prior_ij, prior_xy, ln_g, ln_b, fc_w, fc_b.
// Output (concat): up (B*4096*4096), Mij (B*32768), Mxy (B*32768).
// ---------------------------------------------------------------------------
extern "C" void kernel_launch(void* const* d_in, const int* in_sizes, int n_in,
                              void* d_out, int out_size, void* d_ws, size_t ws_size,
                              hipStream_t stream) {
  const float* context  = (const float*)d_in[0];
  const float* prior_ij = (const float*)d_in[2];
  const float* prior_xy = (const float*)d_in[3];
  const float* ln_g     = (const float*)d_in[4];
  const float* ln_b     = (const float*)d_in[5];
  const float* fc_w     = (const float*)d_in[6];
  const float* fc_b     = (const float*)d_in[7];
  int B = in_sizes[0] / (4096 * 256);

  float* out = (float*)d_out;
  float* up  = out;
  float* Mij = out + (size_t)B * 16777216;
  float* Mxy = Mij + (size_t)B * 32768;

  float*  ws    = (float*)d_ws;
  float2* stats = (float2*)ws;                       // B*4096 float2
  float*  ctxp  = ws + (size_t)B * 8192;             // B*262144 floats
  float*  gpre  = ctxp + (size_t)B * 262144;         // 2*B*32768
  float*  pbuf  = gpre + (size_t)2 * B * 32768;      // 2*B*32768
  float*  gfin  = pbuf + (size_t)2 * B * 32768;      // 2*B*32768

  int rows = B * 4096;
  ln_stats_kernel<<<rows / 8, 256, 0, stream>>>(context, stats, rows);

  int ptotal = B * 262144;
  pool_kernel<<<ptotal / 256, 256, 0, stream>>>(context, stats, ln_g, ln_b, ctxp, ptotal);

  gate_kernel<<<2 * B * 32, 32, 0, stream>>>(ctxp, fc_w, fc_b, prior_ij, prior_xy,
                                             Mij, Mxy, gpre, B);

  int ctotal = 2 * B * 1024;
  cumprod_kernel<<<(ctotal + 255) / 256, 256, 0, stream>>>(gpre, pbuf, ctotal);

  int ftotal = 2 * B * 32768;
  combine_kernel<<<(ftotal + 255) / 256, 256, 0, stream>>>(pbuf, gfin, ftotal);

  expand_kernel<<<B * 64, 256, 0, stream>>>(gfin, up, B);
}
